// MambaHybrid_71038759076014
// MI455X (gfx1250) — matrix operand order repaired and checked by reading, not compiled, not run on hardware
//
#include <hip/hip_runtime.h>
#include <cstdint>
#include <cstddef>

// ---------------------------------------------------------------------------
// MambaHybrid block for MI455X (gfx1250), wave32 + WMMA bf16 + TDM staging.
// - All dense matmuls: v_wmma_f32_16x16x32_bf16 (f32 accumulate), LDS-staged
//   panels moved by the Tensor Data Mover (tensor_load_to_lds, TENSORcnt).
// - Flash attention: WMMA S=QK^T and O+=P*V with online softmax.
// - Selective-scan recurrence in f32 VALU (precision-critical, <2% of FLOPs).
// ---------------------------------------------------------------------------

typedef __bf16 bf16_t;
typedef __attribute__((ext_vector_type(16))) __bf16 v16bf;
typedef __attribute__((ext_vector_type(8)))  float  v8f;
typedef __attribute__((ext_vector_type(4)))  unsigned int ui4;
typedef __attribute__((ext_vector_type(8)))  int  i8v;
typedef __attribute__((ext_vector_type(4)))  int  i4v;

#define Dm     1024
#define DIm    2048
#define HSm    16
#define PSm    128     // DI/HS
#define NSm    64
#define KCm    4
#define CHUNKm 64
#define HAm    16
#define HKVm   4
#define PAm    64      // D/HA
#define Bb     2
#define Lseq   2048
#define NT     (Bb*Lseq)      // 4096 tokens
#define NCc    (Lseq/CHUNKm)  // 32 chunks
#define GQm    (HAm/HKVm)     // 4
#define EPSf   1.1920929e-7f

#if defined(__has_builtin)
#if __has_builtin(__builtin_amdgcn_tensor_load_to_lds) && __has_builtin(__builtin_amdgcn_s_wait_tensorcnt)
#define CDNA5_HAS_TDM 1
#endif
#endif
#ifndef CDNA5_HAS_TDM
#define CDNA5_HAS_TDM 0
#endif

// ---------------- WMMA fragment loaders ------------------------------------
// A-fragment 16x32 bf16, this file's k order (same in B): lanes 0-15 = rows, lane-half selects
// contiguous K 16-block {half*16..half*16+15}.
__device__ inline v16bf load_a_frag(const bf16_t* p) {
    v16bf f;
#pragma unroll
    for (int i = 0; i < 16; ++i) f[i] = p[i];
    return f;
}
// B-matrix 32x16 bf16, this file's k order (same as A): lane = column, lane-half selects contiguous K 16-block.
__device__ inline v16bf load_b_frag(const bf16_t* p, int stride) {
    v16bf f;
#pragma unroll
    for (int i = 0; i < 16; ++i) f[i] = p[i * stride];
    return f;
}
__device__ inline v8f wmma_bf16(v16bf a, v16bf b, v8f c) {
    return __builtin_amdgcn_wmma_f32_16x16x32_bf16(false, a, false, b,
                                                   (short)0, c, false, false);
}

// ---------------- Tensor Data Mover: 2D tile global->LDS --------------------
#if CDNA5_HAS_TDM
__device__ inline unsigned lds_byte_off(const void* p) {
    // generic LDS pointer -> LDS byte address (low 32 bits, ISA 10.2)
    return (unsigned)(unsigned long long)(uintptr_t)p;
}
// Loads `rows` rows of `rowlen` bf16 elements (row stride `stride` elements,
// contiguous into LDS). Descriptor per CDNA5 ISA ch.8 (D# groups 0/1).
__device__ inline void tdm_load_2d_bf16(unsigned lds_off, const void* gsrc,
                                        unsigned rows, unsigned rowlen,
                                        unsigned stride) {
    unsigned long long ga = (unsigned long long)(uintptr_t)gsrc;
    ui4 g0;
    g0[0] = 1u;                                      // count=1 (valid user D#)
    g0[1] = lds_off;                                 // lds_addr (bytes)
    g0[2] = (unsigned)(ga & 0xffffffffull);          // global_addr[31:0]
    g0[3] = (unsigned)((ga >> 32) & 0x01ffffffull)   // global_addr[56:32]
          | (2u << 30);                              // type=2 ("image")
    i8v g1;
    g1[0] = (int)(1u << 16);                         // data_size=1 (2B), mask=0
    g1[1] = (int)((rowlen & 0xffffu) << 16);         // tensor_dim0[15:0]
    g1[2] = (int)(((rowlen >> 16) & 0xffffu) | ((rows & 0xffffu) << 16));
    g1[3] = (int)(((rows >> 16) & 0xffffu) | ((rowlen & 0xffffu) << 16)); // tile_dim0
    g1[4] = (int)(rows & 0xffffu);                   // tile_dim1 (tile_dim2=0)
    g1[5] = (int)stride;                             // tensor_dim0_stride[31:0]
    g1[6] = 0;
    g1[7] = 0;
    i4v gz = {0, 0, 0, 0};
#if defined(__clang_major__) && (__clang_major__ >= 23)
    i8v gz8 = {0, 0, 0, 0, 0, 0, 0, 0};
    __builtin_amdgcn_tensor_load_to_lds(g0, g1, gz, gz, gz8, 0);
#else
    __builtin_amdgcn_tensor_load_to_lds(g0, g1, gz, gz, 0);
#endif
}
#endif

// ---------------- elementwise / conversion kernels --------------------------
__global__ void k_cvt_transpose_bf16(const float* __restrict__ W,
                                     bf16_t* __restrict__ WT, int K, int N) {
    size_t idx = (size_t)blockIdx.x * 256 + threadIdx.x;
    if (idx >= (size_t)K * N) return;
    int k = (int)(idx / N), n = (int)(idx % N);
    WT[(size_t)n * K + k] = (bf16_t)W[idx];
}

__global__ void k_cvt_bf16(const float* __restrict__ X, bf16_t* __restrict__ Y,
                           size_t n) {
    size_t idx = (size_t)blockIdx.x * 256 + threadIdx.x;
    if (idx < n) Y[idx] = (bf16_t)X[idx];
}

__global__ void k_resid_mix(float* __restrict__ dst, const float* __restrict__ xa,
                            const float* __restrict__ xb,
                            const float* __restrict__ mix) {
    size_t idx = (size_t)blockIdx.x * 256 + threadIdx.x;
    if (idx >= (size_t)NT * Dm) return;
    int c = (int)(idx % Dm);
    dst[idx] = mix[c] * xa[idx] + mix[Dm + c] * xb[idx];
}

__global__ void k_rmsnorm_bf16(const float* __restrict__ X,
                               bf16_t* __restrict__ Y) {
    __shared__ float red[256];
    const float* x = X + (size_t)blockIdx.x * Dm;
    float ss = 0.f;
    for (int c = threadIdx.x; c < Dm; c += 256) { float v = x[c]; ss += v * v; }
    red[threadIdx.x] = ss; __syncthreads();
    for (int s = 128; s > 0; s >>= 1) {
        if ((int)threadIdx.x < s) red[threadIdx.x] += red[threadIdx.x + s];
        __syncthreads();
    }
    float r = rsqrtf(red[0] / (float)Dm + EPSf);
    bf16_t* y = Y + (size_t)blockIdx.x * Dm;
    for (int c = threadIdx.x; c < Dm; c += 256) y[c] = (bf16_t)(x[c] * r);
}

__global__ void k_add_scaled(float* __restrict__ dst, const float* __restrict__ base,
                             const float* __restrict__ delta,
                             const float* __restrict__ scale, int Dd, size_t n) {
    size_t idx = (size_t)blockIdx.x * 256 + threadIdx.x;
    if (idx >= n) return;
    int c = (int)(idx % Dd);
    dst[idx] = base[idx] + scale[c] * delta[idx];
}

__global__ void k_conv_silu(const float* __restrict__ vg,
                            const float* __restrict__ cw,
                            const float* __restrict__ cb,
                            float* __restrict__ value) {
    size_t idx = (size_t)blockIdx.x * 256 + threadIdx.x;
    if (idx >= (size_t)NT * DIm) return;
    int c = (int)(idx % DIm);
    size_t t = idx / DIm;
    int l = (int)(t % Lseq);
    size_t brow = t - l;
    float acc = cb[c];
#pragma unroll
    for (int j = 0; j < KCm; ++j) {
        int li = l - (KCm - 1) + j;
        if (li >= 0) acc += cw[c * KCm + j] * vg[(brow + li) * (size_t)(2 * DIm) + c];
    }
    value[idx] = acc / (1.f + expf(-acc));
}

__global__ void k_gate_mul(const float* __restrict__ Y, const float* __restrict__ vg,
                           bf16_t* __restrict__ out) {
    size_t idx = (size_t)blockIdx.x * 256 + threadIdx.x;
    if (idx >= (size_t)NT * DIm) return;
    int c = (int)(idx % DIm);
    size_t t = idx / DIm;
    float g = vg[t * (size_t)(2 * DIm) + DIm + c];
    out[idx] = (bf16_t)(Y[idx] * (g / (1.f + expf(-g))));
}

__global__ void k_relu_sq(const float* __restrict__ H, bf16_t* __restrict__ O,
                          size_t n) {
    size_t idx = (size_t)blockIdx.x * 256 + threadIdx.x;
    if (idx >= n) return;
    float r = fmaxf(H[idx], 0.f);
    O[idx] = (bf16_t)(r * r);
}

// ---------------- bf16 WMMA GEMM with TDM-staged LDS panels -----------------
//   C[M,N] = A[M,K] @ BT[N,K]^T
// Block: 256 thr = 8 waves. Block tile 128(M) x 64(N); wave tile 16x64.
// K staged 64 at a time (2 WMMA sub-steps, 8 WMMAs per wave per stage),
// double-buffered in LDS, moved by the TDM (wave 0 issues; TENSORcnt+barrier).
// Fragments are preloaded per sub-step so WMMAs issue back-to-back.
#define GM_MB 128
#define GM_NB 64
#define GM_KB 64

__global__ void __launch_bounds__(256)
k_gemm_bf16_nt(const bf16_t* __restrict__ A, const bf16_t* __restrict__ BT,
               float* __restrict__ C, int M, int N, int K) {
    __shared__ bf16_t sA[2][GM_MB * GM_KB];   // 32 KB
    __shared__ bf16_t sB[2][GM_NB * GM_KB];   // 16 KB
    const int tid = threadIdx.x;
    const int lane = tid & 31, wave = tid >> 5;
    const int rr = lane & 15, half = lane >> 4;
    const int n0 = blockIdx.x * GM_NB;
    const int m0 = blockIdx.y * GM_MB;
    const int mrow = wave * 16;
    const v8f vzero = {0.f, 0.f, 0.f, 0.f, 0.f, 0.f, 0.f, 0.f};
    v8f acc[4];
#pragma unroll
    for (int i = 0; i < 4; ++i) acc[i] = vzero;
    const int nsteps = K / GM_KB;
    const bf16_t* Ab = A  + (size_t)m0 * K;
    const bf16_t* Bt = BT + (size_t)n0 * K;

#if CDNA5_HAS_TDM
    if (wave == 0) {
        tdm_load_2d_bf16(lds_byte_off(&sA[0][0]), Ab, GM_MB, GM_KB, (unsigned)K);
        tdm_load_2d_bf16(lds_byte_off(&sB[0][0]), Bt, GM_NB, GM_KB, (unsigned)K);
        __builtin_amdgcn_s_wait_tensorcnt((short)0);
    }
    __syncthreads();
    for (int s = 0; s < nsteps; ++s) {
        const int buf = s & 1;
        if (wave == 0 && s + 1 < nsteps) {
            tdm_load_2d_bf16(lds_byte_off(&sA[buf ^ 1][0]),
                             Ab + (size_t)(s + 1) * GM_KB, GM_MB, GM_KB, (unsigned)K);
            tdm_load_2d_bf16(lds_byte_off(&sB[buf ^ 1][0]),
                             Bt + (size_t)(s + 1) * GM_KB, GM_NB, GM_KB, (unsigned)K);
        }
#pragma unroll
        for (int kk = 0; kk < GM_KB; kk += 32) {
            // preload all fragments, then burst 4 WMMAs back-to-back
            v16bf a = load_a_frag(&sA[buf][(mrow + rr) * GM_KB + kk + half * 16]);
            v16bf bfr[4];
#pragma unroll
            for (int ct = 0; ct < 4; ++ct)
                bfr[ct] = load_b_frag(&sB[buf][(ct * 16 + rr) * GM_KB + kk + half * 16], 1);
#pragma unroll
            for (int ct = 0; ct < 4; ++ct)
                acc[ct] = wmma_bf16(a, bfr[ct], acc[ct]);
        }
        if (wave == 0) __builtin_amdgcn_s_wait_tensorcnt((short)0);
        __syncthreads();
    }
#else
    for (int s = 0; s < nsteps; ++s) {
        __syncthreads();
        for (int i = tid; i < GM_MB * GM_KB; i += 256)
            sA[0][i] = Ab[(size_t)(i / GM_KB) * K + s * GM_KB + (i % GM_KB)];
        for (int i = tid; i < GM_NB * GM_KB; i += 256)
            sB[0][i] = Bt[(size_t)(i / GM_KB) * K + s * GM_KB + (i % GM_KB)];
        __syncthreads();
#pragma unroll
        for (int kk = 0; kk < GM_KB; kk += 32) {
            v16bf a = load_a_frag(&sA[0][(mrow + rr) * GM_KB + kk + half * 16]);
            v16bf bfr[4];
#pragma unroll
            for (int ct = 0; ct < 4; ++ct)
                bfr[ct] = load_b_frag(&sB[0][(ct * 16 + rr) * GM_KB + kk + half * 16], 1);
#pragma unroll
            for (int ct = 0; ct < 4; ++ct)
                acc[ct] = wmma_bf16(a, bfr[ct], acc[ct]);
        }
    }
#endif

    const int rbase = half * 8;
#pragma unroll
    for (int ct = 0; ct < 4; ++ct)
#pragma unroll
        for (int r = 0; r < 8; ++r)
            C[(size_t)(m0 + mrow + rbase + r) * N + n0 + ct * 16 + rr] = acc[ct][r];
}

// ---------------- selective scan (f32 VALU) ---------------------------------
__global__ void k_scan_states(const float* __restrict__ value,
                              const float* __restrict__ Bm,
                              const float* __restrict__ A_log,
                              const float* __restrict__ dt_bias,
                              float* __restrict__ states) {
    int bid = blockIdx.x;
    int h = bid % HSm, c = (bid / HSm) % NCc, b = bid / (HSm * NCc);
    float dt  = log1pf(expf(dt_bias[h]));
    float Adt = -expf(A_log[h]) * dt;
    __shared__ float Bs[CHUNKm][NSm];
    __shared__ float xs[CHUNKm][PSm];
    size_t t0 = (size_t)b * Lseq + c * CHUNKm;
    for (int i = threadIdx.x; i < CHUNKm * NSm; i += 256) {
        int t = i / NSm, n = i % NSm;
        float scale = expf(Adt * (float)(CHUNKm - 1 - t)) * dt;
        Bs[t][n] = Bm[(t0 + t) * (size_t)(HSm * NSm) + h * NSm + n] * scale;
    }
    for (int i = threadIdx.x; i < CHUNKm * PSm; i += 256) {
        int t = i / PSm, p = i % PSm;
        xs[t][p] = value[(t0 + t) * (size_t)DIm + h * PSm + p];
    }
    __syncthreads();
    int p  = threadIdx.x & 127;
    int n0 = (threadIdx.x >> 7) * 32;
    float* outp = states + ((size_t)(b * NCc + c) * HSm + h) * (NSm * PSm);
    for (int n = n0; n < n0 + 32; ++n) {
        float s = 0.f;
#pragma unroll
        for (int t = 0; t < CHUNKm; ++t) s += Bs[t][n] * xs[t][p];
        outp[n * PSm + p] = s;
    }
}

__global__ void k_scan_carry(const float* __restrict__ states,
                             float* __restrict__ hcarry,
                             const float* __restrict__ A_log,
                             const float* __restrict__ dt_bias) {
    int h = blockIdx.x % HSm, b = blockIdx.x / HSm;
    float dt   = log1pf(expf(dt_bias[h]));
    float cdec = expf(-expf(A_log[h]) * dt * (float)CHUNKm);
    int p  = threadIdx.x & 127;
    int n0 = (threadIdx.x >> 7) * 32;
    float hreg[32];
#pragma unroll
    for (int i = 0; i < 32; ++i) hreg[i] = 0.f;
    for (int c = 0; c < NCc; ++c) {
        size_t base = ((size_t)(b * NCc + c) * HSm + h) * (NSm * PSm);
#pragma unroll
        for (int i = 0; i < 32; ++i) {
            hcarry[base + (n0 + i) * PSm + p] = hreg[i];
            hreg[i] = hreg[i] * cdec + states[base + (n0 + i) * PSm + p];
        }
    }
}

__global__ void k_scan_out(const float* __restrict__ value,
                           const float* __restrict__ Bm,
                           const float* __restrict__ Cm,
                           const float* __restrict__ hcarry,
                           const float* __restrict__ A_log,
                           const float* __restrict__ dt_bias,
                           const float* __restrict__ Dp_,
                           float* __restrict__ Y) {
    int bid = blockIdx.x;
    int h = bid % HSm, c = (bid / HSm) % NCc, b = bid / (HSm * NCc);
    float dt  = log1pf(expf(dt_bias[h]));
    float Adt = -expf(A_log[h]) * dt;
    float Dp  = Dp_[h];
    __shared__ float Cs[CHUNKm][NSm], Bs2[CHUNKm][NSm], CBL[CHUNKm][CHUNKm];
    __shared__ float pw[CHUNKm + 1];
    size_t t0 = (size_t)b * Lseq + c * CHUNKm;
    for (int i = threadIdx.x; i < CHUNKm * NSm; i += 256) {
        int t = i / NSm, n = i % NSm;
        Cs[t][n]  = Cm[(t0 + t) * (size_t)(HSm * NSm) + h * NSm + n];
        Bs2[t][n] = Bm[(t0 + t) * (size_t)(HSm * NSm) + h * NSm + n];
    }
    if ((int)threadIdx.x <= CHUNKm) pw[threadIdx.x] = expf(Adt * (float)threadIdx.x);
    __syncthreads();
    for (int e = threadIdx.x; e < CHUNKm * CHUNKm; e += 256) {
        int q = e / CHUNKm, k = e % CHUNKm;
        float s = 0.f;
        if (q >= k) {
#pragma unroll
            for (int n = 0; n < NSm; ++n) s += Cs[q][n] * Bs2[k][n];
            s *= pw[q - k] * dt;
            if (q == k) s += Dp;       // fold D*value into the diagonal
        }
        CBL[q][k] = s;
    }
    __syncthreads();
    int p  = threadIdx.x & 127;
    int q0 = (threadIdx.x >> 7) * 32;
    float xv[CHUNKm], hv[NSm];
#pragma unroll
    for (int t = 0; t < CHUNKm; ++t) xv[t] = value[(t0 + t) * (size_t)DIm + h * PSm + p];
    size_t hb = ((size_t)(b * NCc + c) * HSm + h) * (NSm * PSm);
#pragma unroll
    for (int n = 0; n < NSm; ++n) hv[n] = hcarry[hb + n * PSm + p];
    for (int q = q0; q < q0 + 32; ++q) {
        float yd = 0.f, yo = 0.f;
#pragma unroll
        for (int k = 0; k < CHUNKm; ++k) yd += CBL[q][k] * xv[k];
#pragma unroll
        for (int n = 0; n < NSm; ++n) yo += Cs[q][n] * hv[n];
        Y[(t0 + q) * (size_t)DIm + h * PSm + p] = yd + yo * pw[q + 1];
    }
}

// ---------------- attention prep (head RMSNorm + RoPE) ----------------------
__global__ void k_qk_prep(const float* __restrict__ src, bf16_t* __restrict__ dst,
                          int nh, const float* __restrict__ gain, float outscale) {
    int bid = blockIdx.x;                 // t*nh + h
    int h = bid % nh;
    size_t t = (size_t)(bid / nh);
    int l = (int)(t % Lseq), b = (int)(t / Lseq);
    int d = threadIdx.x;                  // 64 threads
    float x = src[t * (size_t)(nh * PAm) + h * PAm + d];
    __shared__ float red[PAm], xn[PAm];
    red[d] = x * x; __syncthreads();
    for (int s = 32; s > 0; s >>= 1) {
        if (d < s) red[d] += red[d + s];
        __syncthreads();
    }
    float r = rsqrtf(red[0] / (float)PAm + EPSf);
    xn[d] = x * r; __syncthreads();
    int a = d & 31;
    float freq = powf(10000.f, -(float)a / 32.f);
    float th = (float)l * freq;
    float cth = cosf(th), sth = sinf(th);
    float o = (d < 32) ? (xn[d] * cth + xn[d + 32] * sth)
                       : (-xn[d - 32] * sth + xn[d] * cth);
    if (gain) o *= gain[h];
    o *= outscale;
    dst[((size_t)(b * nh + h) * Lseq + l) * PAm + d] = (bf16_t)o;
}

__global__ void k_v_prep(const float* __restrict__ src, bf16_t* __restrict__ dst) {
    size_t idx = (size_t)blockIdx.x * 256 + threadIdx.x;
    if (idx >= (size_t)NT * HKVm * PAm) return;
    int d = (int)(idx % PAm);
    size_t r = idx / PAm;
    int h = (int)(r % HKVm);
    size_t t = r / HKVm;
    int l = (int)(t % Lseq), b = (int)(t / Lseq);
    dst[((size_t)(b * HKVm + h) * Lseq + l) * PAm + d] =
        (bf16_t)src[t * (size_t)(HKVm * PAm) + h * PAm + d];
}

// ---------------- flash attention (WMMA, causal, online softmax) ------------
__global__ void k_flash_attn(const bf16_t* __restrict__ Q,
                             const bf16_t* __restrict__ K,
                             const bf16_t* __restrict__ V,
                             bf16_t* __restrict__ O) {
    const int qblk = blockIdx.x;            // 64-row query tile
    const int bh   = blockIdx.y;            // b*HA + h
    const int b = bh / HAm, h = bh % HAm, hk = h / GQm;
    const int tid  = threadIdx.x;
    const int lane = tid & 31, wave = tid >> 5;
    const int rr = lane & 15, half = lane >> 4, rbase = half * 8;
    const int r0 = wave * 16;

    __shared__ bf16_t sQ[64 * 64], sK[64 * 64], sV[64 * 64], sP[64 * 64];
    __shared__ float  sS[64 * 64];
    __shared__ float  mrow[64], lrow[64], arow[64];

    const bf16_t* Qg = Q + ((size_t)(b * HAm + h) * Lseq + qblk * 64) * PAm;
    const bf16_t* Kg = K + ((size_t)(b * HKVm + hk) * Lseq) * PAm;
    const bf16_t* Vg = V + ((size_t)(b * HKVm + hk) * Lseq) * PAm;

    for (int i = tid; i < 64 * 64; i += 128) sQ[i] = Qg[i];
    if (tid < 64) { mrow[tid] = -3.0e38f; lrow[tid] = 0.f; }
    const v8f vzero = {0.f, 0.f, 0.f, 0.f, 0.f, 0.f, 0.f, 0.f};
    v8f acc[4];
#pragma unroll
    for (int i = 0; i < 4; ++i) acc[i] = vzero;
    __syncthreads();

    // Q-fragments are loop-invariant: load once per wave.
    v16bf aq0 = load_a_frag(&sQ[(r0 + rr) * 64 + 0  + half * 16]);
    v16bf aq1 = load_a_frag(&sQ[(r0 + rr) * 64 + 32 + half * 16]);

    for (int kt = 0; kt <= qblk; ++kt) {
        const bf16_t* kg = Kg + (size_t)kt * 64 * PAm;
        const bf16_t* vg = Vg + (size_t)kt * 64 * PAm;
        for (int i = tid; i < 64 * 64; i += 128) { sK[i] = kg[i]; sV[i] = vg[i]; }
        __syncthreads();

        // S tile = Q @ K^T  (each wave: 16 rows x 64 keys)
#pragma unroll
        for (int ct = 0; ct < 4; ++ct) {
            v16bf b0 = load_b_frag(&sK[(ct * 16 + rr) * 64 + 0  + half * 16], 1);
            v16bf b1 = load_b_frag(&sK[(ct * 16 + rr) * 64 + 32 + half * 16], 1);
            v8f s = wmma_bf16(aq0, b0, vzero);
            s = wmma_bf16(aq1, b1, s);
#pragma unroll
            for (int r = 0; r < 8; ++r)
                sS[(r0 + rbase + r) * 64 + ct * 16 + rr] = s[r];
        }
        __syncthreads();

        // online softmax per row
        if (tid < 64) {
            int row = tid, qg = qblk * 64 + row, kb = kt * 64;
            float mold = mrow[row], rmax = mold;
            for (int j = 0; j < 64; ++j) {
                float sij = (kb + j > qg) ? -3.0e38f : sS[row * 64 + j];
                rmax = fmaxf(rmax, sij);
            }
            float alpha = expf(mold - rmax), rsum = 0.f;
            for (int j = 0; j < 64; ++j) {
                float p = (kb + j > qg) ? 0.f : expf(sS[row * 64 + j] - rmax);
                sP[row * 64 + j] = (bf16_t)p;
                rsum += p;
            }
            lrow[row] = lrow[row] * alpha + rsum;
            mrow[row] = rmax;
            arow[row] = alpha;
        }
        __syncthreads();

        // O = O*alpha + P @ V  (P-fragments invariant across the 4 col tiles)
        {
            v16bf ap0 = load_a_frag(&sP[(r0 + rr) * 64 + 0  + half * 16]);
            v16bf ap1 = load_a_frag(&sP[(r0 + rr) * 64 + 32 + half * 16]);
#pragma unroll
            for (int ct = 0; ct < 4; ++ct) {
                v8f a4 = acc[ct];
#pragma unroll
                for (int r = 0; r < 8; ++r) a4[r] *= arow[r0 + rbase + r];
                v16bf bv0 = load_b_frag(&sV[(0  + half * 16) * 64 + ct * 16 + rr], 64);
                v16bf bv1 = load_b_frag(&sV[(32 + half * 16) * 64 + ct * 16 + rr], 64);
                a4 = wmma_bf16(ap0, bv0, a4);
                a4 = wmma_bf16(ap1, bv1, a4);
                acc[ct] = a4;
            }
        }
        __syncthreads();
    }

    // epilogue: divide by row sums, write [b, l, h*PA + d]
#pragma unroll
    for (int ct = 0; ct < 4; ++ct) {
#pragma unroll
        for (int r = 0; r < 8; ++r) {
            int rowl = r0 + rbase + r;
            float val = acc[ct][r] / lrow[rowl];
            size_t t = (size_t)b * Lseq + qblk * 64 + rowl;
            O[t * Dm + h * PAm + ct * 16 + rr] = (bf16_t)val;
        }
    }
}

// ---------------------------------------------------------------------------
extern "C" void kernel_launch(void* const* d_in, const int* in_sizes, int n_in,
                              void* d_out, int out_size, void* d_ws, size_t ws_size,
                              hipStream_t stream) {
    (void)in_sizes; (void)n_in; (void)out_size; (void)ws_size;
    const float* x_in    = (const float*)d_in[0];
    const float* in_proj = (const float*)d_in[1];
    const float* conv_w  = (const float*)d_in[2];
    const float* conv_b  = (const float*)d_in[3];
    const float* B_proj  = (const float*)d_in[4];
    const float* C_proj  = (const float*)d_in[5];
    const float* dt_bias = (const float*)d_in[6];
    const float* A_log   = (const float*)d_in[7];
    const float* D_par   = (const float*)d_in[8];
    const float* out_proj= (const float*)d_in[9];
    const float* mix_m   = (const float*)d_in[10];
    const float* mamba_s = (const float*)d_in[11];
    const float* m_fc    = (const float*)d_in[12];
    const float* m_proj  = (const float*)d_in[13];
    const float* m_mlp_s = (const float*)d_in[14];
    const float* mix_a   = (const float*)d_in[15];
    const float* q_w     = (const float*)d_in[16];
    const float* k_w     = (const float*)d_in[17];
    const float* v_w     = (const float*)d_in[18];
    const float* o_w     = (const float*)d_in[19];
    const float* q_gain  = (const float*)d_in[20];
    const float* attn_s  = (const float*)d_in[21];
    const float* a_fc    = (const float*)d_in[22];
    const float* a_proj  = (const float*)d_in[23];
    const float* a_mlp_s = (const float*)d_in[24];
    float* out = (float*)d_out;

    char* cur = (char*)d_ws;
    auto alloc = [&](size_t bytes) -> char* {
        char* p = cur; cur += (bytes + 255) & ~(size_t)255; return p;
    };

    // f32 activations
    float* xm     = (float*)alloc((size_t)NT * Dm * 4);
    float* xa     = (float*)alloc((size_t)NT * Dm * 4);
    float* big    = (float*)alloc((size_t)NT * 4096 * 4);   // vg / mlp hidden
    float* value  = (float*)alloc((size_t)NT * DIm * 4);
    float* Bmm    = (float*)alloc((size_t)NT * HSm * NSm * 4);
    float* Cmm    = (float*)alloc((size_t)NT * HSm * NSm * 4);
    float* states = (float*)alloc((size_t)Bb * NCc * HSm * NSm * PSm * 4);
    float* hcarry = (float*)alloc((size_t)Bb * NCc * HSm * NSm * PSm * 4);
    float* yscan  = (float*)alloc((size_t)NT * DIm * 4);
    float* tmpD   = (float*)alloc((size_t)NT * Dm * 4);
    // bf16 activations
    bf16_t* n_bf   = (bf16_t*)alloc((size_t)NT * Dm * 2);
    bf16_t* xm_bf  = (bf16_t*)alloc((size_t)NT * Dm * 2);
    bf16_t* hsq_bf = (bf16_t*)alloc((size_t)NT * 4096 * 2);
    // bf16 transposed weights (all fit comfortably in 192MB L2)
    bf16_t* WinT  = (bf16_t*)alloc((size_t)4096 * 1024 * 2);
    bf16_t* WbT   = (bf16_t*)alloc((size_t)1024 * 1024 * 2);
    bf16_t* WcT   = (bf16_t*)alloc((size_t)1024 * 1024 * 2);
    bf16_t* WoutT = (bf16_t*)alloc((size_t)1024 * 2048 * 2);
    bf16_t* WmfT  = (bf16_t*)alloc((size_t)4096 * 1024 * 2);
    bf16_t* WmpT  = (bf16_t*)alloc((size_t)1024 * 4096 * 2);
    bf16_t* WqT   = (bf16_t*)alloc((size_t)1024 * 1024 * 2);
    bf16_t* WkT   = (bf16_t*)alloc((size_t)256  * 1024 * 2);
    bf16_t* WvT   = (bf16_t*)alloc((size_t)256  * 1024 * 2);
    bf16_t* WoT   = (bf16_t*)alloc((size_t)1024 * 1024 * 2);
    bf16_t* WafT  = (bf16_t*)alloc((size_t)4096 * 1024 * 2);
    bf16_t* WapT  = (bf16_t*)alloc((size_t)1024 * 4096 * 2);
    // aliases into dead regions (lifetime-checked):
    bf16_t* ymul_bf = (bf16_t*)states;            // states dead after scan_carry
    float*  qf = hcarry;                          // hcarry dead after scan_out
    float*  kf = hcarry + (size_t)NT * Dm;
    float*  vf = kf     + (size_t)NT * 256;
    bf16_t* q_bf   = (bf16_t*)yscan;              // yscan dead after gate_mul
    bf16_t* k_bf   = q_bf + (size_t)NT * Dm;
    bf16_t* v_bf   = k_bf + (size_t)NT * 256;
    bf16_t* yatt_bf= v_bf + (size_t)NT * 256;

    auto cvtT = [&](const float* W, bf16_t* WT, int K, int N) {
        size_t n = (size_t)K * N;
        k_cvt_transpose_bf16<<<(unsigned)((n + 255) / 256), 256, 0, stream>>>(W, WT, K, N);
    };
    auto gemm = [&](const bf16_t* A, const bf16_t* BT, float* C, int M, int N, int K) {
        dim3 g((unsigned)(N / GM_NB), (unsigned)(M / GM_MB));
        k_gemm_bf16_nt<<<g, 256, 0, stream>>>(A, BT, C, M, N, K);
    };

    // ---- weight conversion (bf16, transposed to [N,K]) ----
    cvtT(in_proj, WinT, 1024, 4096);
    cvtT(B_proj,  WbT,  1024, 1024);
    cvtT(C_proj,  WcT,  1024, 1024);
    cvtT(out_proj,WoutT,2048, 1024);
    cvtT(m_fc,    WmfT, 1024, 4096);
    cvtT(m_proj,  WmpT, 4096, 1024);
    cvtT(q_w,     WqT,  1024, 1024);
    cvtT(k_w,     WkT,  1024, 256);
    cvtT(v_w,     WvT,  1024, 256);
    cvtT(o_w,     WoT,  1024, 1024);
    cvtT(a_fc,    WafT, 1024, 4096);
    cvtT(a_proj,  WapT, 4096, 1024);

    const unsigned gD  = (unsigned)(((size_t)NT * Dm  + 255) / 256);
    const unsigned gDI = (unsigned)(((size_t)NT * DIm + 255) / 256);
    const unsigned g4D = (unsigned)(((size_t)NT * 4096 + 255) / 256);

    // ---- mamba branch ----
    k_resid_mix<<<gD, 256, 0, stream>>>(xm, x_in, x_in, mix_m);
    k_rmsnorm_bf16<<<NT, 256, 0, stream>>>(xm, n_bf);
    k_cvt_bf16<<<gD, 256, 0, stream>>>(xm, xm_bf, (size_t)NT * Dm);
    gemm(n_bf, WinT, big, NT, 4096, 1024);                       // vg
    k_conv_silu<<<gDI, 256, 0, stream>>>(big, conv_w, conv_b, value);
    gemm(xm_bf, WbT, Bmm, NT, 1024, 1024);
    gemm(xm_bf, WcT, Cmm, NT, 1024, 1024);
    k_scan_states<<<Bb * NCc * HSm, 256, 0, stream>>>(value, Bmm, A_log, dt_bias, states);
    k_scan_carry<<<Bb * HSm, 256, 0, stream>>>(states, hcarry, A_log, dt_bias);
    k_scan_out<<<Bb * NCc * HSm, 256, 0, stream>>>(value, Bmm, Cmm, hcarry,
                                                   A_log, dt_bias, D_par, yscan);
    k_gate_mul<<<gDI, 256, 0, stream>>>(yscan, big, ymul_bf);
    gemm(ymul_bf, WoutT, tmpD, NT, 1024, 2048);
    k_add_scaled<<<gD, 256, 0, stream>>>(xm, xm, tmpD, mamba_s, Dm, (size_t)NT * Dm);

    // ---- mamba MLP (relu^2) ----
    k_rmsnorm_bf16<<<NT, 256, 0, stream>>>(xm, n_bf);
    gemm(n_bf, WmfT, big, NT, 4096, 1024);
    k_relu_sq<<<g4D, 256, 0, stream>>>(big, hsq_bf, (size_t)NT * 4096);
    gemm(hsq_bf, WmpT, tmpD, NT, 1024, 4096);
    k_add_scaled<<<gD, 256, 0, stream>>>(xm, xm, tmpD, m_mlp_s, Dm, (size_t)NT * Dm);

    // ---- attention branch ----
    k_resid_mix<<<gD, 256, 0, stream>>>(xa, xm, x_in, mix_a);
    k_rmsnorm_bf16<<<NT, 256, 0, stream>>>(xa, n_bf);
    gemm(n_bf, WqT, qf, NT, 1024, 1024);
    gemm(n_bf, WkT, kf, NT, 256, 1024);
    gemm(n_bf, WvT, vf, NT, 256, 1024);
    k_qk_prep<<<NT * HAm, PAm, 0, stream>>>(qf, q_bf, HAm, q_gain, 0.125f); // 1/sqrt(64)
    k_qk_prep<<<NT * HKVm, PAm, 0, stream>>>(kf, k_bf, HKVm, nullptr, 1.0f);
    k_v_prep<<<(unsigned)(((size_t)NT * HKVm * PAm + 255) / 256), 256, 0, stream>>>(vf, v_bf);
    {
        dim3 g((unsigned)(Lseq / 64), (unsigned)(Bb * HAm));
        k_flash_attn<<<g, 128, 0, stream>>>(q_bf, k_bf, v_bf, yatt_bf);
    }
    gemm(yatt_bf, WoT, tmpD, NT, 1024, 1024);
    k_add_scaled<<<gD, 256, 0, stream>>>(xa, xa, tmpD, attn_s, Dm, (size_t)NT * Dm);

    // ---- attention MLP (relu^2) + final output ----
    k_rmsnorm_bf16<<<NT, 256, 0, stream>>>(xa, n_bf);
    gemm(n_bf, WafT, big, NT, 4096, 1024);
    k_relu_sq<<<g4D, 256, 0, stream>>>(big, hsq_bf, (size_t)NT * 4096);
    gemm(hsq_bf, WapT, tmpD, NT, 1024, 4096);
    k_add_scaled<<<gD, 256, 0, stream>>>(out, xa, tmpD, a_mlp_s, Dm, (size_t)NT * Dm);
}
